// Shared_Endmember_INR_27736898797752
// MI455X (gfx1250) — matrix operand order repaired and checked by reading, not compiled, not run on hardware
//
#include <hip/hip_runtime.h>
#include <math.h>
#include <stdint.h>

typedef __attribute__((ext_vector_type(16))) _Float16 v16h;
typedef __attribute__((ext_vector_type(8)))  float    v8f;
typedef __attribute__((ext_vector_type(4)))  unsigned int u32x4;
typedef __attribute__((ext_vector_type(8)))  int      i32x8;
typedef __attribute__((ext_vector_type(4)))  int      i32x4;

#define LRELU(x) ((x) > 0.f ? (x) : 0.2f * (x))

__device__ __forceinline__ unsigned rfl(unsigned x) {
  return (unsigned)__builtin_amdgcn_readfirstlane((int)x);
}

// ---- WMMA fragment loaders (16x16x32 f16, wave32) -------------------------
// A tile in LDS: row-major [m=16][k=32] halves. Lanes 0-15: row M=lane,
// K=0..15; lanes 16-31: same rows, K=16..31.
__device__ __forceinline__ v16h frag_a_from_lds(const _Float16* tile, int lane) {
  const int m = lane & 15;
  const _Float16* p = tile + m * 32 + ((lane < 16) ? 0 : 16);
  v16h f;
#pragma unroll
  for (int e = 0; e < 16; ++e) f[e] = p[e];
  return f;
}
// B tile staged transposed [n=16][k=32] halves; lanes 0-15 K=0..15, 16-31 K=16..31.
__device__ __forceinline__ v16h frag_b_from_lds(const _Float16* tile, int lane) {
  const int n = lane & 15;
  const _Float16* p = tile + n * 32 + ((lane < 16) ? 0 : 16);
  v16h f;
#pragma unroll
  for (int e = 0; e < 16; ++e) f[e] = p[e];
  return f;
}
// B fragment from an f32 tile (e.g. TDM-loaded raw Y data), cvt on read.
__device__ __forceinline__ v16h frag_b_from_lds_f32(const float* tile, int lane) {
  const int n = lane & 15;
  const float* p = tile + n * 32 + ((lane < 16) ? 0 : 16);
  v16h f;
#pragma unroll
  for (int e = 0; e < 16; ++e) f[e] = (_Float16)p[e];
  return f;
}

// ---- Kernel 1: a = lrelu(conv3x3(Y,191->64)+b1), im2col WMMA GEMM ---------
// Block covers 64 oc x 64 px; tiles staged cooperatively; 2 WMMAs per wave
// per K-chunk. grid (W/64, H, B), block 256 (8 waves).
__global__ __launch_bounds__(256)
void k_conv3x3(const float* __restrict__ Y, const float* __restrict__ w1,
               const float* __restrict__ b1, float* __restrict__ a)
{
  __shared__ _Float16 shA[4][512];   // [oc-tile][m*32+k]
  __shared__ _Float16 shB[4][512];   // [px-subtile][n*32+k]
  const int tid = threadIdx.x, wv = tid >> 5, lane = tid & 31;
  const int octile = wv & 3, phalf = wv >> 2;
  const int oc0 = octile * 16;
  const int x0  = blockIdx.x * 64;
  const int y   = blockIdx.y;
  const int b   = blockIdx.z;
  const float* Yb = Y + (size_t)b * 191 * 128 * 128;

  v8f acc0 = {}, acc1 = {};
  for (int t9 = 0; t9 < 9; ++t9) {
    const int kh = t9 / 3, kw = t9 % 3;
    const int yy = y + kh - 1;
    const bool yok = (unsigned)yy < 128u;
    for (int cc = 0; cc < 6; ++cc) {
      const int c0 = cc * 32;
      // stage A: 64 oc x 32 k weights; 8 k-contiguous elems per thread
#pragma unroll
      for (int j = 0; j < 8; ++j) {
        int e = tid * 8 + j;
        int oc = e >> 5, k = e & 31, c = c0 + k;
        float v = (c < 191) ? w1[((size_t)oc * 191 + c) * 9 + t9] : 0.f;
        shA[oc >> 4][(oc & 15) * 32 + k] = (_Float16)v;
      }
      // stage B: 64 px x 32 ch im2col; 8 x-contiguous elems per thread
#pragma unroll
      for (int j = 0; j < 8; ++j) {
        int e = tid * 8 + j;
        int k = e >> 6, n = e & 63, c = c0 + k;
        int xx = x0 + n + kw - 1;
        float v = 0.f;
        if (c < 191 && yok && (unsigned)xx < 128u)
          v = Yb[((size_t)c * 128 + yy) * 128 + xx];
        shB[n >> 4][(n & 15) * 32 + k] = (_Float16)v;
      }
      // prefetch next channel chunk of this row into cache
      {
        int cp = c0 + 32 + (tid >> 6);
        int xp = x0 + (tid & 63);
        if (cp < 191 && yok)
          __builtin_prefetch(&Yb[((size_t)cp * 128 + yy) * 128 + xp], 0, 1);
      }
      __syncthreads();
      v16h fa  = frag_a_from_lds(shA[octile], lane);
      v16h fb0 = frag_b_from_lds(shB[phalf * 2 + 0], lane);
      v16h fb1 = frag_b_from_lds(shB[phalf * 2 + 1], lane);
      acc0 = __builtin_amdgcn_wmma_f32_16x16x32_f16(false, fa, false, fb0,
                                                    (short)0, acc0, false, false);
      acc1 = __builtin_amdgcn_wmma_f32_16x16x32_f16(false, fa, false, fb1,
                                                    (short)0, acc1, false, false);
      __syncthreads();
    }
  }
  const int n = lane & 15, mhi = (lane < 16) ? 0 : 8;
#pragma unroll
  for (int e = 0; e < 8; ++e) {
    int oc = oc0 + e + mhi;
    float bi = b1[oc];
    size_t base = (((size_t)b * 64 + oc) * 128 + y) * 128;
    float v0 = acc0[e] + bi, v1 = acc1[e] + bi;
    a[base + x0 + (phalf * 2 + 0) * 16 + n] = LRELU(v0);
    a[base + x0 + (phalf * 2 + 1) * 16 + n] = LRELU(v1);
  }
}

// ---- Kernel 2: attention logits = conv1x1(a,64->32)+b2, WMMA GEMM ---------
__global__ __launch_bounds__(256)
void k_logits(const float* __restrict__ a, const float* __restrict__ w2,
              const float* __restrict__ b2, float* __restrict__ logits)
{
  __shared__ _Float16 shA[8][512];
  __shared__ _Float16 shB[8][512];
  const int tid = threadIdx.x, wv = tid >> 5, lane = tid & 31;
  const int m0 = (wv & 1) * 16;
  const int p0 = blockIdx.x * 64 + (wv >> 1) * 16;
  const int b  = blockIdx.z;
  const float* ab = a + (size_t)b * 64 * 16384;

  v8f acc = {};
  for (int cc = 0; cc < 2; ++cc) {
    const int c0 = cc * 32;
#pragma unroll
    for (int i = 0; i < 16; ++i) {
      int idx = lane * 16 + i;
      int m = idx >> 5, k = idx & 31;
      shA[wv][m * 32 + k] = (_Float16)w2[(size_t)(m0 + m) * 64 + (c0 + k)];
    }
#pragma unroll
    for (int i = 0; i < 16; ++i) {
      int idx = lane * 16 + i;
      int n = idx >> 5, k = idx & 31;
      shB[wv][n * 32 + k] = (_Float16)ab[(size_t)(c0 + k) * 16384 + (p0 + n)];
    }
    __syncthreads();
    v16h fa = frag_a_from_lds(shA[wv], lane);
    v16h fb = frag_b_from_lds(shB[wv], lane);
    acc = __builtin_amdgcn_wmma_f32_16x16x32_f16(false, fa, false, fb,
                                                 (short)0, acc, false, false);
    __syncthreads();
  }
  const int n = lane & 15, mhi = (lane < 16) ? 0 : 8;
#pragma unroll
  for (int e = 0; e < 8; ++e) {
    int r = m0 + e + mhi;
    logits[((size_t)b * 32 + r) * 16384 + (p0 + n)] = acc[e] + b2[r];
  }
}

// ---- Kernel 3: per-row softmax stats (max, sum exp) over HW ---------------
__global__ __launch_bounds__(256)
void k_softmax_stats(const float* __restrict__ logits, float* __restrict__ stats)
{
  __shared__ float red[256];
  const int row = blockIdx.x;
  const float* p = logits + (size_t)row * 16384;
  float mx = -1e30f;
  for (int i = threadIdx.x; i < 16384; i += 256) mx = fmaxf(mx, p[i]);
  red[threadIdx.x] = mx; __syncthreads();
  for (int s = 128; s > 0; s >>= 1) {
    if (threadIdx.x < s) red[threadIdx.x] = fmaxf(red[threadIdx.x], red[threadIdx.x + s]);
    __syncthreads();
  }
  mx = red[0]; __syncthreads();
  float sm = 0.f;
  for (int i = threadIdx.x; i < 16384; i += 256) sm += __expf(p[i] - mx);
  red[threadIdx.x] = sm; __syncthreads();
  for (int s = 128; s > 0; s >>= 1) {
    if (threadIdx.x < s) red[threadIdx.x] += red[threadIdx.x + s];
    __syncthreads();
  }
  if (threadIdx.x == 0) { stats[row * 2] = mx; stats[row * 2 + 1] = red[0]; }
}

// ---- Kernel 4: E[b,r,l] = softmax(logits)[b,r,:] . Y[b,l,:], WMMA GEMM ----
// B operand (raw Y tile, 16 rows x 32 f32, row stride 16384) loaded by the
// Tensor Data Mover; one tensor_load_to_lds per wave-pair per K-chunk.
// tensor_dim1 = 191 - l0 gives hardware zero-fill for the ragged last tile.
__global__ __launch_bounds__(256)
void k_E(const float* __restrict__ logits, const float* __restrict__ stats,
         const float* __restrict__ Y, float* __restrict__ E)
{
  __shared__ _Float16 shA[8][512];
  __shared__ float    shBf[4][16][32];   // per k-partition, TDM destination
  __shared__ float    red[8][32][8];
  const int tid = threadIdx.x, wv = tid >> 5, lane = tid & 31;
  const int m0 = (wv & 1) * 16;
  const int part = wv >> 1;
  const int l0 = blockIdx.x * 16;
  const int b  = blockIdx.z;
  const float* lg = logits + (size_t)b * 32 * 16384;
  const float* Yb = Y + (size_t)b * 191 * 16384;

  v8f acc = {};
  const int kbase = part * 4096;
  for (int ch = 0; ch < 128; ++ch) {
    const int k0 = kbase + ch * 32;
    if ((wv & 1) == 0) {
      // Build D# for a 2D tile: 16 rows x 32 cols of f32, row stride 16384.
      const float* gsrc = Yb + (size_t)l0 * 16384 + k0;
      unsigned long long ga = (unsigned long long)(uintptr_t)gsrc;
      unsigned lds_off = (unsigned)(uintptr_t)&shBf[part][0][0];
      unsigned td0 = 16384u - (unsigned)k0;                    // cols remaining
      unsigned td1 = (191 > l0) ? (unsigned)(191 - l0) : 0u;   // rows remaining
      u32x4 g0;
      g0[0] = rfl(1u);                                         // count=1, valid
      g0[1] = rfl(lds_off);                                    // lds_addr
      g0[2] = rfl((unsigned)(ga & 0xffffffffu));               // global_addr lo
      g0[3] = rfl((unsigned)((ga >> 32) & 0x1ffffffu) | (2u << 30)); // hi | type=2
      i32x8 g1;
      g1[0] = (int)rfl(2u << 16);                              // data_size=4B
      g1[1] = (int)rfl((td0 & 0xffffu) << 16);                 // tensor_dim0 lo
      g1[2] = (int)rfl((td0 >> 16) | ((td1 & 0xffffu) << 16)); // dim0 hi | dim1 lo
      g1[3] = (int)rfl((td1 >> 16) | (32u << 16));             // dim1 hi | tile_dim0=32
      g1[4] = (int)rfl(16u);                                   // tile_dim1=16
      g1[5] = (int)rfl(16384u);                                // dim0_stride lo
      g1[6] = 0;                                               // stride hi / dim1_stride
      g1[7] = 0;
      i32x4 gz4 = {0, 0, 0, 0};
      i32x8 gz8 = {0, 0, 0, 0, 0, 0, 0, 0};
      __builtin_amdgcn_tensor_load_to_lds(g0, g1, gz4, gz4, gz8, 0);
    }
    // A = exp(logit - rowmax), staged per wave; 1/sum applied at the end
#pragma unroll
    for (int i = 0; i < 16; ++i) {
      int idx = lane * 16 + i;
      int m = idx >> 5, k = idx & 31;
      int r = m0 + m;
      float mx = stats[(b * 32 + r) * 2];
      shA[wv][m * 32 + k] = (_Float16)__expf(lg[(size_t)r * 16384 + (k0 + k)] - mx);
    }
    if ((wv & 1) == 0) __builtin_amdgcn_s_wait_tensorcnt(0);
    __syncthreads();
    v16h fa = frag_a_from_lds(shA[wv], lane);
    v16h fb = frag_b_from_lds_f32(&shBf[part][0][0], lane);
    acc = __builtin_amdgcn_wmma_f32_16x16x32_f16(false, fa, false, fb,
                                                 (short)0, acc, false, false);
    __syncthreads();
  }
#pragma unroll
  for (int e = 0; e < 8; ++e) red[wv][lane][e] = acc[e];
  __syncthreads();
  if (wv < 2) {
    const int n = lane & 15, mhi = (lane < 16) ? 0 : 8;
#pragma unroll
    for (int e = 0; e < 8; ++e) {
      float s = red[wv][lane][e] + red[wv + 2][lane][e] +
                red[wv + 4][lane][e] + red[wv + 6][lane][e];
      int r = wv * 16 + e + mhi;
      int l = l0 + n;
      if (l < 191)
        E[((size_t)b * 32 + r) * 191 + l] = s / stats[(b * 32 + r) * 2 + 1];
    }
  }
}

// ---- Kernel 5/6: tiny 1-D convs on (2,32/64,191) --------------------------
__global__ void k_conv1d_fe1(const float* __restrict__ E, const float* __restrict__ w,
                             const float* __restrict__ bias, float* __restrict__ out)
{
  int idx = blockIdx.x * blockDim.x + threadIdx.x;
  if (idx >= 2 * 64 * 191) return;
  int l = idx % 191, oc = (idx / 191) % 64, b = idx / (191 * 64);
  float s = bias[oc];
  for (int ic = 0; ic < 32; ++ic) {
    const float* er = E + ((size_t)b * 32 + ic) * 191;
    const float* wr = w + ((size_t)oc * 32 + ic) * 3;
    for (int t = 0; t < 3; ++t) {
      int ll = l + t - 1;
      if (0 <= ll && ll < 191) s += wr[t] * er[ll];
    }
  }
  out[idx] = LRELU(s);
}

__global__ void k_conv1d_fe2(const float* __restrict__ X, const float* __restrict__ w,
                             const float* __restrict__ bias, float* __restrict__ feat)
{
  int idx = blockIdx.x * blockDim.x + threadIdx.x;
  if (idx >= 2 * 64 * 191) return;
  int l = idx % 191, oc = (idx / 191) % 64, b = idx / (191 * 64);
  float s = bias[oc];
  for (int ic = 0; ic < 64; ++ic) {
    const float* xr = X + ((size_t)b * 64 + ic) * 191;
    const float* wr = w + ((size_t)oc * 64 + ic) * 3;
    for (int t = 0; t < 3; ++t) {
      int ll = l + t - 1;
      if (0 <= ll && ll < 191) s += wr[t] * xr[ll];
    }
  }
  feat[idx] = s;   // no activation: feat used raw
}

// ---- Kernel 7: sampling + ray MLP + offsets + aggregation + INR MLP -------
__device__ __forceinline__ float sample1(const float* __restrict__ row, float ixn) {
  float ix = (ixn + 1.f) * 0.5f * 190.f;
  ix = fminf(fmaxf(ix, 0.f), 190.f);
  float x0 = floorf(ix);
  float w  = ix - x0;
  int i0 = (int)x0;
  int i1 = (i0 + 1 < 191) ? i0 + 1 : 190;
  return row[i0] * (1.f - w) + row[i1] * w;
}

__global__ __launch_bounds__(128)
void k_tail(const float* __restrict__ coords, const float* __restrict__ feat,
            const float* __restrict__ rt_w1, const float* __restrict__ rt_b1,
            const float* __restrict__ rt_wr, const float* __restrict__ rt_br,
            const float* __restrict__ rt_wo, const float* __restrict__ rt_bo,
            const float* __restrict__ inr_w1, const float* __restrict__ inr_b1,
            const float* __restrict__ inr_wr1, const float* __restrict__ inr_br1,
            const float* __restrict__ inr_wr2, const float* __restrict__ inr_br2,
            const float* __restrict__ inr_wo, const float* __restrict__ inr_bo,
            const float* __restrict__ temp, float* __restrict__ out)
{
  __shared__ float sh_rin[65], sh_h[64], sh_h2[64], sh_rout[16];
  __shared__ float sh_off[5], sh_w[5], sh_x[77], sh_hid[128], sh_hid2[128];
  const int t = threadIdx.x;
  const int n = blockIdx.x & 511;
  const int b = blockIdx.x >> 9;
  const float cx = coords[n];
  const float* fb = feat + (size_t)b * 64 * 191;

  if (t < 64)  sh_rin[t]  = sample1(fb + t * 191, cx);
  if (t == 64) sh_rin[64] = cx;
  __syncthreads();
  if (t < 64) {
    float s = rt_b1[t];
    for (int i = 0; i < 65; ++i) s += rt_w1[t * 65 + i] * sh_rin[i];
    sh_h[t] = LRELU(s);
  }
  __syncthreads();
  if (t < 64) {
    float s = sh_h[t] + rt_br[t];
    for (int i = 0; i < 64; ++i) s += rt_wr[t * 64 + i] * sh_h[i];
    sh_h2[t] = LRELU(s);
  }
  __syncthreads();
  if (t < 10) {
    float s = rt_bo[t];
    for (int i = 0; i < 64; ++i) s += rt_wo[t * 64 + i] * sh_h2[i];
    sh_rout[t] = s;
  }
  __syncthreads();
  if (t == 0) {
    const float max_off = 6.f / 191.f;
    float mx = sh_rout[5];
    for (int k = 1; k < 5; ++k) mx = fmaxf(mx, sh_rout[5 + k]);
    float sm = 0.f;
    for (int k = 0; k < 5; ++k) { float e = __expf(sh_rout[5 + k] - mx); sh_w[k] = e; sm += e; }
    for (int k = 0; k < 5; ++k) sh_w[k] /= sm;
    for (int k = 0; k < 5; ++k) sh_off[k] = tanhf(sh_rout[k]) * max_off;
  }
  __syncthreads();
  if (t < 64) {
    float s = 0.f;
    for (int k = 0; k < 5; ++k) s += sh_w[k] * sample1(fb + t * 191, cx + sh_off[k]);
    sh_x[t] = s;
  } else if (t < 77) {
    int i = t - 64;
    const float PI = 3.14159265358979323846f;
    if (i == 0)      sh_x[t] = cx;
    else if (i <= 6) sh_x[t] = __sinf(cx * exp2f((float)(i - 1)) * PI);
    else             sh_x[t] = __cosf(cx * exp2f((float)(i - 7)) * PI);
  }
  __syncthreads();
  {
    float s = inr_b1[t];
    for (int i = 0; i < 77; ++i) s += inr_w1[t * 77 + i] * sh_x[i];
    sh_hid[t] = LRELU(s);
  }
  __syncthreads();
  {
    float s = sh_hid[t] + inr_br1[t];
    for (int i = 0; i < 128; ++i) s += inr_wr1[t * 128 + i] * sh_hid[i];
    sh_hid2[t] = LRELU(s);
  }
  __syncthreads();
  {
    float s = sh_hid2[t] + inr_br2[t];
    for (int i = 0; i < 128; ++i) s += inr_wr2[t * 128 + i] * sh_hid2[i];
    sh_hid[t] = LRELU(s);
  }
  __syncthreads();
  if (t < 32) {
    float s = inr_bo[t];
    for (int i = 0; i < 128; ++i) s += inr_wo[t * 128 + i] * sh_hid[i];
    float tc = fminf(fmaxf(temp[0], 0.5f), 10.f);
    out[((size_t)b * 512 + n) * 32 + t] = 1.f / (1.f + __expf(-s / tc));
  }
}

// ---------------------------------------------------------------------------
extern "C" void kernel_launch(void* const* d_in, const int* in_sizes, int n_in,
                              void* d_out, int out_size, void* d_ws, size_t ws_size,
                              hipStream_t stream)
{
  const float* coords  = (const float*)d_in[0];
  const float* Y       = (const float*)d_in[1];
  const float* ext_w1  = (const float*)d_in[2];
  const float* ext_b1  = (const float*)d_in[3];
  const float* ext_w2  = (const float*)d_in[4];
  const float* ext_b2  = (const float*)d_in[5];
  const float* fe_w1   = (const float*)d_in[6];
  const float* fe_b1   = (const float*)d_in[7];
  const float* fe_w2   = (const float*)d_in[8];
  const float* fe_b2   = (const float*)d_in[9];
  const float* rt_w1   = (const float*)d_in[10];
  const float* rt_b1   = (const float*)d_in[11];
  const float* rt_wr   = (const float*)d_in[12];
  const float* rt_br   = (const float*)d_in[13];
  const float* rt_wo   = (const float*)d_in[14];
  const float* rt_bo   = (const float*)d_in[15];
  const float* inr_w1  = (const float*)d_in[16];
  const float* inr_b1  = (const float*)d_in[17];
  const float* inr_wr1 = (const float*)d_in[18];
  const float* inr_br1 = (const float*)d_in[19];
  const float* inr_wr2 = (const float*)d_in[20];
  const float* inr_br2 = (const float*)d_in[21];
  const float* inr_wo  = (const float*)d_in[22];
  const float* inr_bo  = (const float*)d_in[23];
  const float* temp    = (const float*)d_in[24];

  float* ws     = (float*)d_ws;
  float* a      = ws;                       // 2*64*16384  = 2097152 f
  float* logits = a + 2 * 64 * 16384;       // 2*32*16384  = 1048576 f
  float* stats  = logits + 2 * 32 * 16384;  // 64*2        = 128 f
  float* E      = stats + 128;              // 2*32*191    = 12224 f
  float* tmp1   = E + 2 * 32 * 191;         // 2*64*191    = 24448 f
  float* feat   = tmp1 + 2 * 64 * 191;      // 2*64*191    = 24448 f

  k_conv3x3      <<<dim3(2, 128, 2), 256, 0, stream>>>(Y, ext_w1, ext_b1, a);
  k_logits       <<<dim3(256, 1, 2), 256, 0, stream>>>(a, ext_w2, ext_b2, logits);
  k_softmax_stats<<<64, 256, 0, stream>>>(logits, stats);
  k_E            <<<dim3(12, 1, 2), 256, 0, stream>>>(logits, stats, Y, E);
  k_conv1d_fe1   <<<(2 * 64 * 191 + 255) / 256, 256, 0, stream>>>(E, fe_w1, fe_b1, tmp1);
  k_conv1d_fe2   <<<(2 * 64 * 191 + 255) / 256, 256, 0, stream>>>(tmp1, fe_w2, fe_b2, feat);
  k_tail<<<1024, 128, 0, stream>>>(coords, feat, rt_w1, rt_b1, rt_wr, rt_br,
                                   rt_wo, rt_bo, inr_w1, inr_b1, inr_wr1, inr_br1,
                                   inr_wr2, inr_br2, inr_wo, inr_bo, temp,
                                   (float*)d_out);
}
